// NeuralExposureControl_25975962206438
// MI455X (gfx1250) — matrix operand order repaired and checked by reading, not compiled, not run on hardware
//
#include <hip/hip_runtime.h>
#include <hip/hip_bf16.h>
#include <math.h>

// ---------------- problem constants ----------------
#define H_   1200
#define W_   1920
#define B_   4
#define HW_  (H_*W_)
#define NB_  256
#define NCROP 59          // total crops; only every 4th column of the 236 is used
#define NCOL  59          // used network columns (stride-4 of 236)
#define MIN_CLIP_F (1.0f/1048575.0f)

typedef __attribute__((ext_vector_type(16))) _Float16 v16h;
typedef __attribute__((ext_vector_type(8)))  _Float16 v8h;
typedef __attribute__((ext_vector_type(8)))  float    v8f;

// ---------------- mask predicates (pure function of H,W; mirrors reference) --
__device__ __host__ __forceinline__ bool mask_row_f(int y) {
    int ry = 4 + (y * (H_ - 4)) / H_;
    if (ry & 1) return false;
    int g = ry >> 1;
    for (int by = 0; by < 21; ++by) {
        int base = (int)(by * (H_ / 42.0));
        int t = g - 4 - base;
        if (t >= 0 && t < 25) return true;
    }
    return false;
}
__device__ __host__ __forceinline__ bool mask_col_f(int x) {
    int rx = 4 + (x * (W_ - 4)) / W_;
    if (rx & 1) return false;
    int h = rx >> 1;
    for (int bx = 0; bx < 21; ++bx) {
        int base = (int)(bx * (W_ / 42.0));
        int t = h - 5 - base;
        if (t >= 0 && t < 40) return true;
    }
    return false;
}

// ---------------- exposure merge (exact multiple of 1/256 in [0,4095]) ------
__device__ __forceinline__ float merge_val(float l, float m, float h) {
    float low  = rintf(l * 4095.0f);
    float mid  = rintf(m * 4095.0f);
    float high = rintf(h * 4095.0f);
    float r = 0.0f;
    if      (mid == 4095.0f)                  r = low;
    else if (mid <= 255.0f)                   r = high * (1.0f/256.0f);
    else if (mid >= 256.0f && mid <= 4094.0f) r = mid  * (1.0f/16.0f);
    return r;
}

// ---------------- kernels ----------------
__global__ void k_zero_u32(unsigned* p, int n) {
    int i = blockIdx.x * blockDim.x + threadIdx.x;
    if (i < n) p[i] = 0u;
}

// Pass 1: merge all pixels -> m_raw; 4 px/thread, b128 traffic (roofline term)
__global__ void k_merge4(const float* __restrict__ img, float* __restrict__ mraw) {
    int i = blockIdx.x * blockDim.x + threadIdx.x;
    const int perB = HW_ / 4;
    if (i >= B_ * perB) return;
    int b = i / perB, p = (i - b * perB) * 4;
    const float* base = img + (size_t)b * 3 * HW_;
    __builtin_prefetch(base + p + 4096, 0, 1);               // global_prefetch_b8
    float4 lo = *(const float4*)(base + p);
    float4 mi = *(const float4*)(base + HW_ + p);
    float4 hi = *(const float4*)(base + 2*HW_ + p);
    float4 r;
    r.x = merge_val(lo.x, mi.x, hi.x);
    r.y = merge_val(lo.y, mi.y, hi.y);
    r.z = merge_val(lo.z, mi.z, hi.z);
    r.w = merge_val(lo.w, mi.w, hi.w);
    *(float4*)(mraw + (size_t)b * HW_ + p) = r;
}

// Pass 2a: coarse 4096-bin histogram of q=m*256 over masked pixels
__global__ void k_coarse_hist(const float* __restrict__ mraw, unsigned* __restrict__ coarse) {
    int i = blockIdx.x * blockDim.x + threadIdx.x;
    if (i >= B_ * HW_) return;
    int b = i / HW_, p = i - b * HW_;
    int y = p / W_, x = p - y * W_;
    if (!mask_row_f(y) || !mask_col_f(x)) return;
    int q = (int)(mraw[i] * 256.0f + 0.5f);      // exact: value*256 is integer <= 2^20
    atomicAdd(&coarse[b * 4096 + (q >> 8)], 1u);
}

// Pass 2b: find coarse bin containing order statistic k-1
__global__ void k_select_bin(const unsigned* __restrict__ coarse, unsigned* __restrict__ minfo, int kth) {
    int b = blockIdx.x;
    if (threadIdx.x != 0) return;
    unsigned target = (unsigned)(kth - 1);
    unsigned cum = 0;
    for (int c = 0; c < 4096; ++c) {
        unsigned cnt = coarse[b * 4096 + c];
        if (cum + cnt > target) { minfo[b*2] = (unsigned)c; minfo[b*2+1] = target - cum; return; }
        cum += cnt;
    }
    minfo[b*2] = 4095u; minfo[b*2+1] = 0u;
}

// Pass 2c: fine 256-bin histogram inside the selected coarse bin
__global__ void k_fine_hist(const float* __restrict__ mraw, const unsigned* __restrict__ minfo,
                            unsigned* __restrict__ fine) {
    int i = blockIdx.x * blockDim.x + threadIdx.x;
    if (i >= B_ * HW_) return;
    int b = i / HW_, p = i - b * HW_;
    int y = p / W_, x = p - y * W_;
    if (!mask_row_f(y) || !mask_col_f(x)) return;
    int q = (int)(mraw[i] * 256.0f + 0.5f);
    if ((unsigned)(q >> 8) != minfo[b*2]) return;
    atomicAdd(&fine[b * NB_ + (q & 255)], 1u);
}

// Pass 2d: exact median -> scale[b] = 1/(1024*max(median,MIN_CLIP))
__global__ void k_median_scale(const unsigned* __restrict__ fine, const unsigned* __restrict__ minfo,
                               float* __restrict__ scale) {
    int b = threadIdx.x;
    if (b >= B_) return;
    unsigned rank = minfo[b*2+1], cum = 0;
    int sub = 255;
    for (int s = 0; s < 256; ++s) {
        cum += fine[b * NB_ + s];
        if (cum > rank) { sub = s; break; }
    }
    float med = (float)((int)minfo[b*2] * 256 + sub) * (1.0f/256.0f);
    med = fmaxf(med, MIN_CLIP_F);
    scale[b] = 1.0f / (1024.0f * med);
}

// Pass 3: log-histograms, only for the 59 columns conv1 actually reads.
// LDS-privatized per block, merged with global atomics.
__global__ void __launch_bounds__(256)
k_crop_hist(const float* __restrict__ mraw, const float* __restrict__ scale,
            unsigned* __restrict__ chist) {
    __shared__ unsigned sh[NB_];
    const int j = blockIdx.y;               // network column 0..58
    const int g = 4 * j;                    // original column in 0..235
    const int b = g / NCROP, c = g % NCROP; // (batch, crop)
    int sc, ci, cj;
    if (c == 0)       { sc = 1; ci = 0;            cj = 0; }
    else if (c < 10)  { sc = 3; ci = (c-1)/3;      cj = (c-1)%3; }
    else              { sc = 7; ci = (c-10)/7;     cj = (c-10)%7; }
    int y1 = (int)((double)ci     / sc * H_);
    int y2 = (int)((double)(ci+1) / sc * H_);
    int x1 = (int)((double)cj     / sc * W_);
    int x2 = (int)((double)(cj+1) / sc * W_);
    x1 += (x1 & 1);
    y1 = y1 + (y1 & 1) + (1 - 2 * (y1 & 1));   // force odd, as in reference
    int ny = (y2 > y1) ? (y2 - y1 + 1) / 2 : 0;
    int nx = (x2 > x1) ? (x2 - x1 + 1) / 2 : 0;
    int npts = ny * nx;

    sh[threadIdx.x] = 0u;
    __syncthreads();

    const float logmc = logf(MIN_CLIP_F);
    const float sb = scale[b];
    const float* mb = mraw + (size_t)b * HW_;
    for (int p = blockIdx.x * blockDim.x + threadIdx.x; p < npts; p += gridDim.x * blockDim.x) {
        int py = p / nx, px = p - py * nx;
        float m = mb[(y1 + 2*py) * W_ + (x1 + 2*px)];
        float v = fminf(m * sb, 1.0f);
        v = fmaxf(v, MIN_CLIP_F);
        float t = (logmc - logf(v)) / logmc;
        int bin = (int)(t * 256.0f);
        bin = min(max(bin, 0), 255);
        atomicAdd(&sh[bin], 1u);
    }
    __syncthreads();
    if (sh[threadIdx.x]) atomicAdd(&chist[j * NB_ + threadIdx.x], sh[threadIdx.x]);
}

// Pass 4: normalize histograms -> Xc[256][59] (conv1 input, already column-subsampled)
__global__ void __launch_bounds__(256)
k_hist_norm(const unsigned* __restrict__ chist, float* __restrict__ Xc) {
    __shared__ float sh[NB_];
    int j = blockIdx.x, t = threadIdx.x;
    float cnt = (float)chist[j * NB_ + t];
    sh[t] = cnt;
    __syncthreads();
    for (int s = 128; s > 0; s >>= 1) {
        if (t < s) sh[t] += sh[t + s];
        __syncthreads();
    }
    float total = fmaxf(sh[0], 0.0001f);
    Xc[t * NCOL + j] = cnt / total * 256.0f;
}

// conv1: K=4, done on VALU; fused bias+relu, output f16 [128][64][59]
__global__ void k_conv1(const float* __restrict__ Xc, const float* __restrict__ w1,
                        const float* __restrict__ b1, _Float16* __restrict__ out1) {
    int idx = blockIdx.x * blockDim.x + threadIdx.x;
    if (idx >= 128 * 64 * 59) return;
    int j = idx % 59, i = (idx / 59) % 64, o = idx / (59 * 64);
    float acc = b1[o];
    #pragma unroll
    for (int k = 0; k < 4; ++k)
        acc += w1[o * 4 + k] * Xc[(4*i + k) * NCOL + j];
    out1[idx] = (_Float16)fmaxf(acc, 0.0f);
}

__global__ void k_f32_to_f16(const float* __restrict__ s, _Float16* __restrict__ d, int n) {
    int i = blockIdx.x * blockDim.x + threadIdx.x;
    if (i < n) d[i] = (_Float16)s[i];
}

// im2col packs, writing B TRANSPOSED (N x K) so every lane's fragment is K-contiguous
__global__ void k_pack_b2(const _Float16* __restrict__ a1, _Float16* __restrict__ Bt) {
    int idx = blockIdx.x * blockDim.x + threadIdx.x;       // 240*512
    if (idx >= 240 * 512) return;
    int n = idx / 512, kap = idx % 512;
    int ci = kap >> 2, kk = kap & 3, i = n / 15, j = n % 15;
    Bt[idx] = a1[ci * (64*59) + (4*i + kk) * 59 + 4*j];
}
__global__ void k_pack_b3(const _Float16* __restrict__ a2, _Float16* __restrict__ Bt) {
    int idx = blockIdx.x * blockDim.x + threadIdx.x;       // 16*1024
    if (idx >= 16 * 1024) return;
    int n = idx / 1024, kap = idx % 1024;
    int ci = kap >> 2, kk = kap & 3, i = n >> 2, j = n & 3;
    Bt[idx] = a2[ci * 240 + (4*i + kk) * 15 + 4*j];
}
__global__ void k_pack_b4(const _Float16* __restrict__ a3, _Float16* __restrict__ Bt) {
    int idx = blockIdx.x * blockDim.x + threadIdx.x;       // 16*2048 (N padded 1->16)
    if (idx >= 16 * 2048) return;
    int n = idx / 2048, kap = idx % 2048;
    _Float16 v = (_Float16)0.0f;
    if (n == 0) {
        int ci = kap >> 2, kk = kap & 3;
        v = a3[ci * 16 + kk * 4];  // act3[ci][h=kk][w=0]
    }
    Bt[idx] = v;
}

// Generic 16x16-tiled WMMA GEMM: C = relu(A[MxK] @ Bt[NxK]^T + bias).
// One wave32 per tile. Tiles staged to LDS with async b128 copies (ASYNCcnt),
// fragments read back as two ds_load_b128 per operand in contiguous k order, k = 16h + s:
//   A lane: K chunks [akb..akb+7] and [akb+8..akb+15]   (akb = 0|16)
//   B lane: K chunks [bkb..bkb+7] and [bkb+8..bkb+15]   (bkb = 0|16)
__global__ void __launch_bounds__(32)
k_wmma_gemm(const _Float16* __restrict__ A, const _Float16* __restrict__ Bt,
            const float* __restrict__ bias, _Float16* __restrict__ C,
            int M, int N, int K, int do_relu) {
    __shared__ __align__(16) _Float16 shA[16 * 32];
    __shared__ __align__(16) _Float16 shB[16 * 32];
    const int lane = threadIdx.x;
    const int tn = blockIdx.x, tm = blockIdx.y;
    const int hi = (lane >= 16);
    const int r  = lane & 15;
    const int akb = hi ? 16 : 0;
    const int bkb = hi ? 16 : 0;
    v8f acc = {};
    for (int k0 = 0; k0 < K; k0 += 32) {
        // stage 16x32 A tile and 16x32 B tile: 32 lanes x 2 shots x 16B each
        #pragma unroll
        for (int s = 0; s < 2; ++s) {
            int cch = s * 32 + lane;               // 0..63
            int row = cch >> 2, ch = (cch & 3) * 8;
            const _Float16* ga = A  + (size_t)(tm * 16 + row) * K + k0 + ch;
            const _Float16* gb = Bt + (size_t)(tn * 16 + row) * K + k0 + ch;
            unsigned la = (unsigned)(uintptr_t)&shA[row * 32 + ch];
            unsigned lb = (unsigned)(uintptr_t)&shB[row * 32 + ch];
            asm volatile("global_load_async_to_lds_b128 %0, %1, off"
                         :: "v"(la), "v"((unsigned long long)(uintptr_t)ga) : "memory");
            asm volatile("global_load_async_to_lds_b128 %0, %1, off"
                         :: "v"(lb), "v"((unsigned long long)(uintptr_t)gb) : "memory");
        }
        asm volatile("s_wait_asynccnt 0x0" ::: "memory");

        v8h a0 = *(const v8h*)&shA[r * 32 + akb];
        v8h a1 = *(const v8h*)&shA[r * 32 + akb + 8];
        v8h b0 = *(const v8h*)&shB[r * 32 + bkb];
        v8h b1 = *(const v8h*)&shB[r * 32 + bkb + 8];
        v16h a = __builtin_shufflevector(a0, a1, 0,1,2,3,4,5,6,7,8,9,10,11,12,13,14,15);
        v16h b = __builtin_shufflevector(b0, b1, 0,1,2,3,4,5,6,7,8,9,10,11,12,13,14,15);
        acc = __builtin_amdgcn_wmma_f32_16x16x32_f16(false, a, false, b,
                                                     (short)0, acc, false, false);
    }
    #pragma unroll
    for (int rr = 0; rr < 8; ++rr) {
        int m = rr + (hi ? 8 : 0);
        float v = acc[rr] + bias[tm * 16 + m];
        if (do_relu) v = fmaxf(v, 0.0f);
        C[(size_t)(tm * 16 + m) * N + (tn * 16 + r)] = (_Float16)v;
    }
}

// Head: conv5 (16x1024 GEMV) + conv6 (1x16) + 2*(sigmoid(3x)-0.5)*ln4 -> exp
__global__ void __launch_bounds__(32)
k_head(const _Float16* __restrict__ o4, const float* __restrict__ w5, const float* __restrict__ b5,
       const float* __restrict__ w6, const float* __restrict__ b6, float* __restrict__ out) {
    __shared__ float sh[16];
    int t = threadIdx.x;
    if (t < 16) {
        float s = b5[t];
        for (int ci = 0; ci < 1024; ++ci)
            s += w5[t * 1024 + ci] * (float)o4[ci * 16];   // column 0 of padded N
        sh[t] = fmaxf(s, 0.0f);
    }
    __syncthreads();
    if (t == 0) {
        float s = b6[0];
        #pragma unroll
        for (int o = 0; o < 16; ++o) s += w6[o] * sh[o];
        float sig = 1.0f / (1.0f + expf(-3.0f * s));
        float y = 2.0f * (sig - 0.5f) * 1.3862943611198906f;   // ln(4)
        out[0] = expf(y);
    }
}

// ---------------- host launcher ----------------
extern "C" void kernel_launch(void* const* d_in, const int* in_sizes, int n_in,
                              void* d_out, int out_size, void* d_ws, size_t ws_size,
                              hipStream_t stream) {
    const float* img = (const float*)d_in[0];
    const float* w1  = (const float*)d_in[1];  const float* b1 = (const float*)d_in[2];
    const float* w2  = (const float*)d_in[3];  const float* b2 = (const float*)d_in[4];
    const float* w3  = (const float*)d_in[5];  const float* b3 = (const float*)d_in[6];
    const float* w4  = (const float*)d_in[7];  const float* b4 = (const float*)d_in[8];
    const float* w5  = (const float*)d_in[9];  const float* b5 = (const float*)d_in[10];
    const float* w6  = (const float*)d_in[11]; const float* b6 = (const float*)d_in[12];
    float* out = (float*)d_out;

    // median rank k (host-computable: mask is a pure function of H,W)
    int nrows = 0, ncols = 0;
    for (int y = 0; y < H_; ++y) if (mask_row_f(y)) ++nrows;
    for (int x = 0; x < W_; ++x) if (mask_col_f(x)) ++ncols;
    long long N = (long long)nrows * ncols;
    int kth = (int)((N / 2) > 0 ? (N / 2) : 1);

    // ---- workspace carving (~44 MB, all f16 regions 16B aligned) ----
    unsigned* coarse = (unsigned*)d_ws;                 // B*4096
    unsigned* fine   = coarse + B_ * 4096;              // B*256
    unsigned* chist  = fine   + B_ * NB_;               // 59*256
    unsigned* minfo  = chist  + NCOL * NB_;             // B*2
    float* scale = (float*)(minfo + B_ * 2);            // B
    float* Xc    = scale + B_;                          // 256*59
    float* mraw  = Xc + NB_ * NCOL;                     // B*H*W
    _Float16* out1 = (_Float16*)(mraw + (size_t)B_ * HW_); // 128*64*59
    _Float16* w2h = out1 + 128*64*59;                   // 256*512
    _Float16* w3h = w2h + 256*512;                      // 512*1024
    _Float16* w4h = w3h + 512*1024;                     // 1024*2048
    _Float16* Bt2 = w4h + 1024*2048;                    // 240*512 (transposed)
    _Float16* Bt3 = Bt2 + 240*512;                      // 16*1024 (transposed)
    _Float16* Bt4 = Bt3 + 16*1024;                      // 16*2048 (transposed)
    _Float16* o2  = Bt4 + 16*2048;                      // 256*240
    _Float16* o3  = o2 + 256*240;                       // 512*16
    _Float16* o4  = o3 + 512*16;                        // 1024*16

    const int nzero = B_*4096 + B_*NB_ + NCOL*NB_ + B_*2;
    k_zero_u32<<<(nzero + 255)/256, 256, 0, stream>>>(coarse, nzero);

    const int npx = B_ * HW_;
    k_merge4<<<(npx/4 + 255)/256, 256, 0, stream>>>(img, mraw);
    k_coarse_hist<<<(npx + 255)/256, 256, 0, stream>>>(mraw, coarse);
    k_select_bin<<<B_, 32, 0, stream>>>(coarse, minfo, kth);
    k_fine_hist<<<(npx + 255)/256, 256, 0, stream>>>(mraw, minfo, fine);
    k_median_scale<<<1, 32, 0, stream>>>(fine, minfo, scale);

    k_crop_hist<<<dim3(96, NCOL), 256, 0, stream>>>(mraw, scale, chist);
    k_hist_norm<<<NCOL, 256, 0, stream>>>(chist, Xc);

    k_conv1<<<(128*64*59 + 255)/256, 256, 0, stream>>>(Xc, w1, b1, out1);

    k_f32_to_f16<<<(256*512   + 255)/256, 256, 0, stream>>>(w2, w2h, 256*512);
    k_f32_to_f16<<<(512*1024  + 255)/256, 256, 0, stream>>>(w3, w3h, 512*1024);
    k_f32_to_f16<<<(1024*2048 + 255)/256, 256, 0, stream>>>(w4, w4h, 1024*2048);

    k_pack_b2<<<(240*512 + 255)/256, 256, 0, stream>>>(out1, Bt2);
    k_wmma_gemm<<<dim3(240/16, 256/16), 32, 0, stream>>>(w2h, Bt2, b2, o2, 256, 240, 512, 1);
    k_pack_b3<<<(16*1024 + 255)/256, 256, 0, stream>>>(o2, Bt3);
    k_wmma_gemm<<<dim3(16/16, 512/16), 32, 0, stream>>>(w3h, Bt3, b3, o3, 512, 16, 1024, 1);
    k_pack_b4<<<(16*2048 + 255)/256, 256, 0, stream>>>(o3, Bt4);
    k_wmma_gemm<<<dim3(16/16, 1024/16), 32, 0, stream>>>(w4h, Bt4, b4, o4, 1024, 16, 2048, 1);

    k_head<<<1, 32, 0, stream>>>(o4, w5, b5, w6, b6, out);

    (void)in_sizes; (void)n_in; (void)out_size; (void)ws_size;
}
